// TransformerDecoder_90993177133807
// MI455X (gfx1250) — matrix operand order repaired and checked by reading, not compiled, not run on hardware
//
#include <hip/hip_runtime.h>
#include <math.h>

typedef __attribute__((ext_vector_type(16))) __bf16 v16bf;
typedef __attribute__((ext_vector_type(8)))  float  v8f;
typedef __attribute__((ext_vector_type(4)))  int    v4i;
typedef __attribute__((address_space(1))) v4i gv4i;   // global
typedef __attribute__((address_space(3))) v4i lv4i;   // LDS

union FragBf { v16bf v; uint4 u[2]; };
union AccF   { v8f   v; float f[8]; };

__device__ __forceinline__ unsigned short f2bf(float f) {
    union { float f; unsigned u; } c; c.f = f;
    unsigned r = c.u + 0x7FFFu + ((c.u >> 16) & 1u);   // round-to-nearest-even
    return (unsigned short)(r >> 16);
}
__device__ __forceinline__ float bf2f(unsigned short h) {
    union { unsigned u; float f; } c; c.u = ((unsigned)h) << 16;
    return c.f;
}
__device__ __forceinline__ unsigned pk2(float a, float b) {
    return (unsigned)f2bf(a) | ((unsigned)f2bf(b) << 16);
}
__device__ __forceinline__ v8f wmma_bf16(const FragBf& a, const FragBf& b, v8f c) {
    return __builtin_amdgcn_wmma_f32_16x16x32_bf16(false, a.v, false, b.v,
                                                   (short)0, c, false, false);
}

// ---- CDNA5 async global->LDS (ASYNCcnt) ----
__device__ __forceinline__ void async_b128(unsigned short* lds, const unsigned short* g) {
#if __has_builtin(__builtin_amdgcn_global_load_async_to_lds_b128)
    __builtin_amdgcn_global_load_async_to_lds_b128((gv4i*)g, (lv4i*)lds, 0, 0);
#else
    unsigned loff = (unsigned)(unsigned long long)lds;   // generic LDS ptr low 32 bits = LDS offset
    asm volatile("global_load_async_to_lds_b128 %0, %1, off" :: "v"(loff), "v"(g) : "memory");
#endif
}
__device__ __forceinline__ void wait_async_8() {
#if __has_builtin(__builtin_amdgcn_s_wait_asynccnt)
    __builtin_amdgcn_s_wait_asynccnt(8);
#else
    asm volatile("s_wait_asynccnt 0x8" ::: "memory");
#endif
}
__device__ __forceinline__ void wait_async_0() {
#if __has_builtin(__builtin_amdgcn_s_wait_asynccnt)
    __builtin_amdgcn_s_wait_asynccnt(0);
#else
    asm volatile("s_wait_asynccnt 0x0" ::: "memory");
#endif
}

enum { EPI_BIAS = 0, EPI_SOFTSIGN = 1, EPI_GELU = 2, EPI_RES = 3 };

// Issue one 128x64(bf16) A tile + one 128x64 B tile as async copies (8 wave-instructions).
// OOB rows are clamped (uniform ASYNCcnt); garbage only reaches masked-out columns.
__device__ __forceinline__ void stage_tiles(const unsigned short* __restrict__ Ab, int lda, int m0, int M,
                                            const unsigned short* __restrict__ Bb, int ldb, int n0, int N,
                                            int k0, unsigned short* sA, unsigned short* sB, int tid)
{
#pragma unroll
    for (int j = 0; j < 4; ++j) {
        int f = tid + j * 256;          // 1024 x B128 = 128 rows x 64 bf16
        int r = f >> 3, kc = (f & 7) << 3;
        int gm = m0 + r; if (gm >= M) gm = M - 1;
        async_b128(sA + r * 64 + kc, Ab + (long)gm * lda + k0 + kc);
        int gn = n0 + r; if (gn >= N) gn = N - 1;
        async_b128(sB + r * 64 + kc, Bb + (long)gn * ldb + k0 + kc);
    }
}

// C[M,N] = A[M,K](bf16) x B[N,K]^T(bf16) + bias (+res), fp32 accum.
// Tile 128x128x64, 256 threads = 8 waves, double-buffered async LDS staging.
template<int EPI, bool OUTBF>
__global__ __launch_bounds__(256)
void gemm_bf16(const unsigned short* __restrict__ A, long aZ, int lda,
               const unsigned short* __restrict__ Bm, long bZ, int ldb,
               const float* __restrict__ bias, long biasZ,
               const float* __restrict__ res,
               void* __restrict__ Cp, long cZ, int ldc,
               int M, int N, int K, const int* __restrict__ zsel)
{
    __shared__ unsigned short sA[2][128 * 64];   // 16KB each
    __shared__ unsigned short sB[2][128 * 64];

    const int z  = blockIdx.z;
    const int bz = zsel ? zsel[z] : z;
    const unsigned short* Ab = A  + (long)z * aZ;
    const unsigned short* Bb = Bm + (long)z * bZ;
    const float* biasb = bias + (long)bz * biasZ;
    const float* resb  = res ? (res + (long)z * cZ) : nullptr;

    const int tid  = threadIdx.x;
    const int m0   = blockIdx.y * 128;
    const int n0   = blockIdx.x * 128;
    const int wave = tid >> 5, lane = tid & 31;
    const int half = lane >> 4, l16 = lane & 15;
    const int wm   = (wave >> 2) * 64;
    const int wn   = (wave & 3) * 32;

    AccF acc[4][2];
#pragma unroll
    for (int mi = 0; mi < 4; ++mi)
#pragma unroll
        for (int ni = 0; ni < 2; ++ni)
#pragma unroll
            for (int r = 0; r < 8; ++r) acc[mi][ni].f[r] = 0.f;

    const int ksteps = K >> 6;
    stage_tiles(Ab, lda, m0, M, Bb, ldb, n0, N, 0, sA[0], sB[0], tid);

    for (int ks = 0; ks < ksteps; ++ks) {
        const int cur = ks & 1;
        if (ks + 1 < ksteps) {
            stage_tiles(Ab, lda, m0, M, Bb, ldb, n0, N, (ks + 1) << 6,
                        sA[cur ^ 1], sB[cur ^ 1], tid);
            wait_async_8();            // current stage landed
        } else {
            wait_async_0();
        }
        __syncthreads();

        const unsigned short* cA = sA[cur];
        const unsigned short* cB = sB[cur];
#pragma unroll
        for (int kk = 0; kk < 64; kk += 32) {
            FragBf afr[4], bfr[2];
#pragma unroll
            for (int mi = 0; mi < 4; ++mi) {
                const uint4* p = (const uint4*)(cA + (wm + mi * 16 + l16) * 64 + kk + half * 8);
                afr[mi].u[0] = p[0];     // K {0..7 | 8..15}
                afr[mi].u[1] = p[2];     // K {16..23 | 24..31}
            }
#pragma unroll
            for (int ni = 0; ni < 2; ++ni) {
                const uint4* p = (const uint4*)(cB + (wn + ni * 16 + l16) * 64 + kk + half * 8);
                bfr[ni].u[0] = p[0];     // K {0..7 | 8..15}
                bfr[ni].u[1] = p[2];     // K {16..23 | 24..31}
            }
#pragma unroll
            for (int mi = 0; mi < 4; ++mi)
#pragma unroll
                for (int ni = 0; ni < 2; ++ni)
                    acc[mi][ni].v = wmma_bf16(afr[mi], bfr[ni], acc[mi][ni].v);
        }
        __syncthreads();   // all waves done reading before this buffer is overwritten
    }

#pragma unroll
    for (int mi = 0; mi < 4; ++mi)
#pragma unroll
        for (int ni = 0; ni < 2; ++ni)
#pragma unroll
            for (int r = 0; r < 8; ++r) {
                int row = m0 + wm + mi * 16 + r + half * 8;
                int col = n0 + wn + ni * 16 + l16;
                if (row < M && col < N) {
                    float v = acc[mi][ni].f[r] + biasb[col];
                    if (EPI == EPI_SOFTSIGN)  v = v / (1.0f + fabsf(v));
                    else if (EPI == EPI_GELU) v = 0.5f * v * (1.0f + erff(v * 0.70710678118f));
                    else if (EPI == EPI_RES)  v += resb[(long)row * ldc + col];
                    if (OUTBF) ((unsigned short*)Cp + (long)z * cZ)[(long)row * ldc + col] = f2bf(v);
                    else       ((float*)Cp + (long)z * cZ)[(long)row * ldc + col] = v;
                }
            }
}

// Flash attention on bf16 qkv. grid (qtile=4, h=8, b=16), block 256 = 8 waves.
__global__ __launch_bounds__(256)
void flash_attn(const unsigned short* __restrict__ qkv, unsigned short* __restrict__ O)
{
    __shared__ unsigned short sQ[128 * 64];      // [t][d]
    __shared__ unsigned short sK[64 * 64];       // [t][d]
    __shared__ unsigned short sVT[64 * 64];      // [d][t]
    __shared__ float          sS[8 * 16 * 64];   // per-wave scores
    __shared__ unsigned short sP[8 * 16 * 64];   // per-wave P (bf16)
    __shared__ float sM[128], sL[128], sAl[128];

    const int qt = blockIdx.x, h = blockIdx.y, b = blockIdx.z;
    const int tid = threadIdx.x, wave = tid >> 5, lane = tid & 31;
    const int half = lane >> 4, l16 = lane & 15;
    const float scale = 0.125f;

    const long qbase = ((long)(b * 512 + qt * 128)) * 1536 + h * 64;
    const long kbase = ((long)(b * 512)) * 1536 + 512 + h * 64;
    const long vbase = ((long)(b * 512)) * 1536 + 1024 + h * 64;

    // async Q tile (4 wave-instructions; completes at first wait_async_0)
#pragma unroll
    for (int j = 0; j < 4; ++j) {
        int f = tid + j * 256;                   // 1024 x B128 = 128x64 bf16
        int t = f >> 3, d = (f & 7) << 3;
        async_b128(sQ + t * 64 + d, qkv + qbase + (long)t * 1536 + d);
    }
    if (tid < 128) { sM[tid] = -3.0e38f; sL[tid] = 0.f; }

    AccF o[4];
#pragma unroll
    for (int ni = 0; ni < 4; ++ni)
#pragma unroll
        for (int r = 0; r < 8; ++r) o[ni].f[r] = 0.f;

    for (int kt = 0; kt < 8; ++kt) {
        __syncthreads();                         // sK/sVT free for reuse
        // async K tile (2 wave-instructions)
#pragma unroll
        for (int j = 0; j < 2; ++j) {
            int f = tid + j * 256;               // 512 x B128 = 64x64 bf16
            int t = f >> 3, d = (f & 7) << 3;
            async_b128(sK + t * 64 + d, qkv + kbase + (long)(kt * 64 + t) * 1536 + d);
        }
        // V transposed via VGPRs
#pragma unroll
        for (int j = 0; j < 8; ++j) {
            int f = tid + j * 256;               // 2048 x u32 = 64x64 bf16
            int t = f >> 5, d2 = (f & 31) << 1;
            unsigned v = *(const unsigned*)(qkv + vbase + (long)(kt * 64 + t) * 1536 + d2);
            sVT[d2 * 64 + t]       = (unsigned short)(v & 0xffffu);
            sVT[(d2 + 1) * 64 + t] = (unsigned short)(v >> 16);
        }
        wait_async_0();
        __syncthreads();

        // S = Q Kt (16x64 per wave)
        AccF s[4];
#pragma unroll
        for (int ni = 0; ni < 4; ++ni)
#pragma unroll
            for (int r = 0; r < 8; ++r) s[ni].f[r] = 0.f;
#pragma unroll
        for (int kk = 0; kk < 64; kk += 32) {
            FragBf a;
            const uint4* pa = (const uint4*)(sQ + (wave * 16 + l16) * 64 + kk + half * 8);
            a.u[0] = pa[0]; a.u[1] = pa[2];
#pragma unroll
            for (int ni = 0; ni < 4; ++ni) {
                FragBf bb;
                const uint4* pb = (const uint4*)(sK + (ni * 16 + l16) * 64 + kk + half * 8);
                bb.u[0] = pb[0]; bb.u[1] = pb[2];
                s[ni].v = wmma_bf16(a, bb, s[ni].v);
            }
        }
        float* myS = sS + wave * 1024;
#pragma unroll
        for (int ni = 0; ni < 4; ++ni)
#pragma unroll
            for (int r = 0; r < 8; ++r)
                myS[(r + half * 8) * 64 + ni * 16 + l16] = s[ni].f[r] * scale;
        __syncthreads();

        if (lane < 16) {
            int row = wave * 16 + lane;
            float* Sr = sS + wave * 1024 + lane * 64;
            unsigned short* Pr = sP + wave * 1024 + lane * 64;
            float mo = sM[row], mx = mo;
            for (int j2 = 0; j2 < 64; ++j2) mx = fmaxf(mx, Sr[j2]);
            float l = 0.f;
            for (int j2 = 0; j2 < 64; ++j2) {
                float p = expf(Sr[j2] - mx);
                l += p; Pr[j2] = f2bf(p);
            }
            float alpha = expf(mo - mx);
            sAl[row] = alpha; sM[row] = mx;
            sL[row] = sL[row] * alpha + l;
        }
        __syncthreads();

#pragma unroll
        for (int ni = 0; ni < 4; ++ni)
#pragma unroll
            for (int r = 0; r < 8; ++r)
                o[ni].f[r] *= sAl[wave * 16 + r + half * 8];
#pragma unroll
        for (int kk = 0; kk < 64; kk += 32) {
            FragBf a;
            const uint4* pa = (const uint4*)(sP + wave * 1024 + l16 * 64 + kk + half * 8);
            a.u[0] = pa[0]; a.u[1] = pa[2];
#pragma unroll
            for (int ni = 0; ni < 4; ++ni) {
                FragBf bb;
                const uint4* pb = (const uint4*)(sVT + (ni * 16 + l16) * 64 + kk + half * 8);
                bb.u[0] = pb[0]; bb.u[1] = pb[2];
                o[ni].v = wmma_bf16(a, bb, o[ni].v);
            }
        }
    }
    __syncthreads();

    const long orow0 = (long)(b * 512 + qt * 128);
#pragma unroll
    for (int ni = 0; ni < 4; ++ni)
#pragma unroll
        for (int r = 0; r < 8; ++r) {
            int rloc = wave * 16 + r + half * 8;
            int col  = h * 64 + ni * 16 + l16;
            O[(orow0 + rloc) * 512 + col] = f2bf(o[ni].f[r] / sL[rloc]);
        }
}

// wave-per-row LayerNorm: fp32 in, bf16 out
__global__ __launch_bounds__(256)
void layernorm(const float* __restrict__ X, const float* __restrict__ g,
               const float* __restrict__ bb, unsigned short* __restrict__ Y, int nrows)
{
    int wave = threadIdx.x >> 5, lane = threadIdx.x & 31;
    int row = blockIdx.x * 8 + wave;
    if (row >= nrows) return;
    const float* x = X + (long)row * 512;
    float s = 0.f, q = 0.f, vals[16];
#pragma unroll
    for (int i = 0; i < 16; ++i) {
        float v = x[lane + i * 32];
        vals[i] = v; s += v; q += v * v;
    }
#pragma unroll
    for (int off = 16; off; off >>= 1) {
        s += __shfl_xor(s, off, 32);
        q += __shfl_xor(q, off, 32);
    }
    float mean = s * (1.f / 512.f);
    float var  = q * (1.f / 512.f) - mean * mean;
    float rs   = rsqrtf(var + 1e-6f);
    unsigned short* y = Y + (long)row * 512;
#pragma unroll
    for (int i = 0; i < 16; ++i) {
        int c = lane + i * 32;
        y[c] = f2bf((vals[i] - mean) * rs * g[c] + bb[c]);
    }
}

// RoPE on bf16 q/k halves of qkv
__global__ void rope_kernel(unsigned short* __restrict__ qkv,
                            const float* __restrict__ cs, const float* __restrict__ sn)
{
    int idx = blockIdx.x * 256 + threadIdx.x;    // 2^22 threads
    int j  = idx & 31;
    int hh = (idx >> 5) & 7;
    int p  = (idx >> 8) & 511;
    int bb = (idx >> 17) & 15;
    int qk = idx >> 21;
    long base = ((long)(bb * 512 + p)) * 1536 + qk * 512 + hh * 64 + 2 * j;
    unsigned v = *(const unsigned*)(qkv + base);
    float c = cs[p * 32 + j], s = sn[p * 32 + j];
    float a = bf2f((unsigned short)(v & 0xffffu));
    float d = bf2f((unsigned short)(v >> 16));
    *(unsigned*)(qkv + base) = pk2(a * c - d * s, a * s + d * c);
}

// fp32 -> bf16 bulk convert
__global__ void cvt_bf16(const float* __restrict__ in, unsigned short* __restrict__ out, int n)
{
    int i = blockIdx.x * 256 + threadIdx.x;
    if (i < n) out[i] = f2bf(in[i]);
}

// Gather the 16 used day matrices, transpose (K,N)->(N,K), convert to bf16
__global__ void day_gather_t(const float* __restrict__ day_w, const int* __restrict__ idx,
                             unsigned short* __restrict__ out)
{
    int z = blockIdx.y;
    int t = blockIdx.x * 256 + threadIdx.x;      // 262144 per z
    int n = t >> 9, k = t & 511;
    out[(long)z * 262144 + t] = f2bf(day_w[(long)idx[z] * 262144 + (long)k * 512 + n]);
}

extern "C" void kernel_launch(void* const* d_in, const int* in_sizes, int n_in,
                              void* d_out, int out_size, void* d_ws, size_t ws_size,
                              hipStream_t stream)
{
    (void)in_sizes; (void)n_in; (void)out_size; (void)ws_size;
    const float* x_in    = (const float*)d_in[0];
    const int*   day_idx = (const int*)d_in[1];
    const float* rcos    = (const float*)d_in[2];
    const float* rsin    = (const float*)d_in[3];
    const float* day_w   = (const float*)d_in[4];
    const float* day_b   = (const float*)d_in[5];
    const float* patch_w = (const float*)d_in[6];
    const float* patch_b = (const float*)d_in[7];
    const float* ln1_g   = (const float*)d_in[8];
    const float* ln1_b   = (const float*)d_in[9];
    const float* ln2_g   = (const float*)d_in[10];
    const float* ln2_b   = (const float*)d_in[11];
    const float* qkv_w   = (const float*)d_in[12];
    const float* qkv_b   = (const float*)d_in[13];
    const float* proj_w  = (const float*)d_in[14];
    const float* proj_b  = (const float*)d_in[15];
    const float* fc1_w   = (const float*)d_in[16];
    const float* fc1_b   = (const float*)d_in[17];
    const float* fc2_w   = (const float*)d_in[18];
    const float* fc2_b   = (const float*)d_in[19];
    const float* lnf_g   = (const float*)d_in[20];
    const float* lnf_b   = (const float*)d_in[21];
    const float* head_w  = (const float*)d_in[22];
    const float* head_b  = (const float*)d_in[23];
    float* out = (float*)d_out;

    char* w = (char*)d_ws;
    auto alloc = [&](size_t bytes) { char* p = w; w += (bytes + 255) & ~(size_t)255; return p; };
    unsigned short* Xb      = (unsigned short*)alloc(16777216ULL * 2);   // x in bf16
    unsigned short* dayT    = (unsigned short*)alloc(16ULL * 512 * 512 * 2);
    unsigned short* qkvWb   = (unsigned short*)alloc(8ULL * 1536 * 512 * 2);
    unsigned short* projWb  = (unsigned short*)alloc(8ULL * 512 * 512 * 2);
    unsigned short* fc1Wb   = (unsigned short*)alloc(8ULL * 2048 * 512 * 2);
    unsigned short* fc2Wb   = (unsigned short*)alloc(8ULL * 512 * 2048 * 2);
    unsigned short* patchWb = (unsigned short*)alloc(512ULL * 2048 * 2);
    unsigned short* headWb  = (unsigned short*)alloc(41ULL * 512 * 2);
    unsigned short* Hb      = (unsigned short*)alloc(8192ULL * 2048 * 2); // day out / fc1 out
    float*          X       = (float*)alloc(8192ULL * 512 * 4);           // residual stream
    unsigned short* LNb     = (unsigned short*)alloc(8192ULL * 512 * 2);
    unsigned short* QKVb    = (unsigned short*)alloc(8192ULL * 1536 * 2);
    unsigned short* Ob      = (unsigned short*)alloc(8192ULL * 512 * 2);

    // ---- one-time conversions (bf16 weights + input) ----
    cvt_bf16<<<65536, 256, 0, stream>>>(x_in, Xb, 16777216);
    day_gather_t<<<dim3(1024, 16), 256, 0, stream>>>(day_w, day_idx, dayT);
    cvt_bf16<<<24576, 256, 0, stream>>>(qkv_w,  qkvWb,  6291456);
    cvt_bf16<<<8192,  256, 0, stream>>>(proj_w, projWb, 2097152);
    cvt_bf16<<<32768, 256, 0, stream>>>(fc1_w,  fc1Wb,  8388608);
    cvt_bf16<<<32768, 256, 0, stream>>>(fc2_w,  fc2Wb,  8388608);
    cvt_bf16<<<4096,  256, 0, stream>>>(patch_w, patchWb, 1048576);
    cvt_bf16<<<82,    256, 0, stream>>>(head_w, headWb, 20992);

    // ---- day-specific linear + softsign ----
    gemm_bf16<EPI_SOFTSIGN, true><<<dim3(4, 16, 16), 256, 0, stream>>>(
        Xb, 2048L * 512, 512, dayT, 262144L, 512, day_b, 512, nullptr,
        Hb, 2048L * 512, 512, 2048, 512, 512, day_idx);
    // ---- patch embed ----
    gemm_bf16<EPI_BIAS, false><<<dim3(4, 64, 1), 256, 0, stream>>>(
        Hb, 0, 2048, patchWb, 0, 2048, patch_b, 0, nullptr,
        X, 0, 512, 8192, 512, 2048, nullptr);

    for (int i = 0; i < 8; ++i) {
        layernorm<<<1024, 256, 0, stream>>>(X, ln1_g + i * 512, ln1_b + i * 512, LNb, 8192);
        gemm_bf16<EPI_BIAS, true><<<dim3(12, 64, 1), 256, 0, stream>>>(
            LNb, 0, 512, qkvWb + (long)i * 1536 * 512, 0, 512, qkv_b + i * 1536, 0, nullptr,
            QKVb, 0, 1536, 8192, 1536, 512, nullptr);
        rope_kernel<<<16384, 256, 0, stream>>>(QKVb, rcos, rsin);
        flash_attn<<<dim3(4, 8, 16), 256, 0, stream>>>(QKVb, Ob);
        gemm_bf16<EPI_RES, false><<<dim3(4, 64, 1), 256, 0, stream>>>(
            Ob, 0, 512, projWb + (long)i * 512 * 512, 0, 512, proj_b + i * 512, 0, X,
            X, 0, 512, 8192, 512, 512, nullptr);
        layernorm<<<1024, 256, 0, stream>>>(X, ln2_g + i * 512, ln2_b + i * 512, LNb, 8192);
        gemm_bf16<EPI_GELU, true><<<dim3(16, 64, 1), 256, 0, stream>>>(
            LNb, 0, 512, fc1Wb + (long)i * 2048 * 512, 0, 512, fc1_b + i * 2048, 0, nullptr,
            Hb, 0, 2048, 8192, 2048, 512, nullptr);
        gemm_bf16<EPI_RES, false><<<dim3(4, 64, 1), 256, 0, stream>>>(
            Hb, 0, 2048, fc2Wb + (long)i * 512 * 2048, 0, 2048, fc2_b + i * 512, 0, X,
            X, 0, 512, 8192, 512, 2048, nullptr);
    }

    layernorm<<<1024, 256, 0, stream>>>(X, lnf_g, lnf_b, LNb, 8192);
    gemm_bf16<EPI_BIAS, false><<<dim3(1, 64, 1), 256, 0, stream>>>(
        LNb, 0, 512, headWb, 0, 512, head_b, 0, nullptr,
        out, 0, 41, 8192, 41, 512, nullptr);
}
